// WhisperSelfAttentionStepTVM_24962349925058
// MI455X (gfx1250) — matrix operand order repaired and checked by reading, not compiled, not run on hardware
//
#include <hip/hip_runtime.h>
#include <hip/hip_bf16.h>
#include <cstdint>

// ---------------------------------------------------------------------------
// WhisperSelfAttentionStep decode kernel for gfx1250 (MI455X, wave32, WMMA)
// B=64, H=20, P=448, D=64, E=1280
// ---------------------------------------------------------------------------

typedef __attribute__((ext_vector_type(16))) _Float16 v16h;
typedef __attribute__((ext_vector_type(8)))  float    v8f;
typedef __attribute__((ext_vector_type(2)))  int      v2i;

#define EDIM   1280
#define BATCH  64
#define HEADS  20
#define PAST   448
#define HDIM   64

// gfx1250 async global->LDS path (ASYNCcnt-tracked), guarded so compile never
// regresses if this toolchain doesn't declare the builtins.
#if defined(__gfx1250__) &&                                            \
    __has_builtin(__builtin_amdgcn_global_load_async_to_lds_b64) &&    \
    __has_builtin(__builtin_amdgcn_s_wait_asynccnt)
#define USE_ASYNC_LDS 1
#else
#define USE_ASYNC_LDS 0
#endif

#if USE_ASYNC_LDS
typedef __attribute__((address_space(1))) v2i* gptr_v2i;
typedef __attribute__((address_space(3))) v2i* lptr_v2i;

__device__ __forceinline__ void async_row8(const float* src, float* lds_dst)
{
    // 8 bytes per lane: one wave moves one 256B K-row per issue.
    __builtin_amdgcn_global_load_async_to_lds_b64(
        (gptr_v2i)(uintptr_t)src,
        (lptr_v2i)(uintptr_t)lds_dst,
        0, 0);
}
#endif

// ---------------------------------------------------------------------------
// WMMA 16x16 tile GEMM core:  out[m0:16, n0:16] = X[m0:16, :K] @ W[n0:16, :K]^T
// A fragment (16-bit A 16x32, contiguous k, same order as B):
//   lane<16 : e -> K=e      (M = lane&15)
//   lane>=16: e -> K=e+16
// B fragment (row-striped):
//   lane<16 : e -> K=e      (N = lane&15)
//   lane>=16: e -> K=e+16
// C/D: VGPR r, lanes 0-15: M=r, N=lane; lanes 16-31: M=r+8, N=lane-16.
// ---------------------------------------------------------------------------
__device__ __forceinline__ v8f wmma_tile_xwt(const float* __restrict__ X,
                                             const float* __restrict__ W,
                                             int m0, int n0, int K, int lane)
{
    const int rowA = m0 + (lane & 15);
    const int rowB = n0 + (lane & 15);
    const int khiA = (lane < 16) ? 0 : 16;
    const int khiB = (lane < 16) ? 0 : 16;

    v8f c = {};
    for (int kb = 0; kb < K; kb += 32) {
        const float* xa = X + (size_t)rowA * K + kb + khiA;
        const float* wb = W + (size_t)rowB * K + kb + khiB;
        v16h a, b;
#pragma unroll
        for (int e = 0; e < 16; ++e)
            a[e] = (_Float16)xa[e];
#pragma unroll
        for (int e = 0; e < 16; ++e)
            b[e] = (_Float16)wb[e];
        c = __builtin_amdgcn_wmma_f32_16x16x32_f16(
                /*neg_a=*/false, a, /*neg_b=*/false, b,
                /*c_mod=*/(short)0, c, /*reuse_a=*/false, /*reuse_b=*/false);
    }
    return c;
}

// ---------------------------------------------------------------------------
// Kernel 1: fused QKV projection.
//   grid.x = 240 (80 n-tiles x 3 projections), grid.y = 4 (m-tiles), block = 32
// ---------------------------------------------------------------------------
__global__ void __launch_bounds__(32)
qkv_gemm_wmma(const float* __restrict__ X,
              const float* __restrict__ Wq, const float* __restrict__ bq,
              const float* __restrict__ Wk,
              const float* __restrict__ Wv, const float* __restrict__ bv,
              float* __restrict__ q_ws,
              float* __restrict__ new_k,
              float* __restrict__ new_v)
{
    const int lane = threadIdx.x;
    const int tn   = blockIdx.x;        // 0..239
    const int proj = tn / 80;           // 0=q, 1=k, 2=v
    const int n0   = (tn % 80) * 16;
    const int m0   = blockIdx.y * 16;

    const float* W = (proj == 0) ? Wq : (proj == 1) ? Wk : Wv;
    const v8f c = wmma_tile_xwt(X, W, m0, n0, EDIM, lane);

    const int ncol  = n0 + (lane & 15);
    const int mbase = m0 + ((lane < 16) ? 0 : 8);

    if (proj == 0) {
        const float bias  = bq[ncol];
        const float scale = 0.125f;      // 64^-0.5
#pragma unroll
        for (int r = 0; r < 8; ++r)
            q_ws[(size_t)(mbase + r) * EDIM + ncol] = (c[r] + bias) * scale;
    } else if (proj == 1) {
#pragma unroll
        for (int r = 0; r < 8; ++r)
            new_k[(size_t)(mbase + r) * EDIM + ncol] = c[r];
    } else {
        const float bias = bv[ncol];
#pragma unroll
        for (int r = 0; r < 8; ++r)
            new_v[(size_t)(mbase + r) * EDIM + ncol] = c[r] + bias;
    }
}

// ---------------------------------------------------------------------------
// Kernel 2: attention over 449 keys per (b,h). One block per (b,h), 8 waves.
// K-pass: async global->LDS double buffer (256B row per wave per issue,
//         ASYNCcnt-tracked) + wave32 shuffle reduction for q.k.
// softmax: block reduce in LDS.
// V-pass: direct float2/lane streaming accumulate.
// ---------------------------------------------------------------------------
__global__ void __launch_bounds__(256)
attn_decode(const float* __restrict__ q_ws,
            const float* __restrict__ past_k,
            const float* __restrict__ past_v,
            const float* __restrict__ keep_mask,   // [448]
            const float* __restrict__ new_k,       // [64*20*64]
            const float* __restrict__ new_v,
            float* __restrict__ ctx_ws)            // [64 x 1280] merged
{
    const int bh   = blockIdx.x;        // b*20 + h
    const int tid  = threadIdx.x;
    const int wave = tid >> 5;
    const int lane = tid & 31;

    __shared__ float s_scores[449];
    __shared__ float s_red[256];
    __shared__ float s_part[8][64];
#if USE_ASYNC_LDS
    __shared__ float s_kbuf[8][2][64];  // per-wave double buffer (4KB)
#endif

    const float* qp = q_ws + (size_t)bh * HDIM;    // q_ws[b*1280 + h*64]
    const float2 q2 = *(const float2*)(qp + 2 * lane);

    const float* kbase = past_k + (size_t)bh * PAST * HDIM;
    const float* vbase = past_v + (size_t)bh * PAST * HDIM;
    const float* knew  = new_k + (size_t)bh * HDIM;
    const float* vnew  = new_v + (size_t)bh * HDIM;

    // ---- scores: q . k_p + mask_p ----
#if USE_ASYNC_LDS
    {
        // prime buffer 0 with this wave's first row
        const float* r0 = (wave < PAST) ? (kbase + (size_t)wave * HDIM) : knew;
        async_row8(r0 + 2 * lane, &s_kbuf[wave][0][2 * lane]);
        int buf = 0;
        for (int p = wave; p < PAST + 1; p += 8) {
            const int pn = p + 8;
            if (pn < PAST + 1) {
                const float* rn = (pn < PAST) ? (kbase + (size_t)pn * HDIM) : knew;
                async_row8(rn + 2 * lane, &s_kbuf[wave][buf ^ 1][2 * lane]);
                __builtin_amdgcn_s_wait_asynccnt(1);   // oldest (row p) landed
            } else {
                __builtin_amdgcn_s_wait_asynccnt(0);
            }
            const float2 kk = *(const float2*)&s_kbuf[wave][buf][2 * lane];
            buf ^= 1;
            float part = q2.x * kk.x + q2.y * kk.y;
#pragma unroll
            for (int off = 16; off >= 1; off >>= 1)
                part += __shfl_xor(part, off, 32);
            if (lane == 0)
                s_scores[p] = part + ((p < PAST) ? keep_mask[p] : 0.0f);
        }
    }
#else
    for (int p = wave; p < PAST + 1; p += 8) {
        if (p + 8 < PAST)
            __builtin_prefetch(kbase + (size_t)(p + 8) * HDIM + 2 * lane, 0, 0);
        const float* kr = (p < PAST) ? (kbase + (size_t)p * HDIM) : knew;
        const float2 kk = *(const float2*)(kr + 2 * lane);
        float part = q2.x * kk.x + q2.y * kk.y;
#pragma unroll
        for (int off = 16; off >= 1; off >>= 1)
            part += __shfl_xor(part, off, 32);
        if (lane == 0)
            s_scores[p] = part + ((p < PAST) ? keep_mask[p] : 0.0f);
    }
#endif
    __syncthreads();

    // ---- softmax: block max ----
    float mx = -3.0e38f;
    for (int p = tid; p < PAST + 1; p += 256)
        mx = fmaxf(mx, s_scores[p]);
    s_red[tid] = mx;
    __syncthreads();
#pragma unroll
    for (int s = 128; s > 0; s >>= 1) {
        if (tid < s) s_red[tid] = fmaxf(s_red[tid], s_red[tid + s]);
        __syncthreads();
    }
    const float gmax = s_red[0];
    __syncthreads();

    // ---- exp + block sum ----
    float psum = 0.0f;
    for (int p = tid; p < PAST + 1; p += 256) {
        const float e = __expf(s_scores[p] - gmax);
        s_scores[p] = e;
        psum += e;
    }
    s_red[tid] = psum;
    __syncthreads();
#pragma unroll
    for (int s = 128; s > 0; s >>= 1) {
        if (tid < s) s_red[tid] += s_red[tid + s];
        __syncthreads();
    }
    const float inv = 1.0f / s_red[0];

    // ---- ctx = sum_p w_p * v_p (unnormalized, fold 1/sum at the end) ----
    float2 acc = make_float2(0.0f, 0.0f);
    for (int p = wave; p < PAST + 1; p += 8) {
        if (p + 8 < PAST)
            __builtin_prefetch(vbase + (size_t)(p + 8) * HDIM + 2 * lane, 0, 0);
        const float* vr = (p < PAST) ? (vbase + (size_t)p * HDIM) : vnew;
        const float2 vv = *(const float2*)(vr + 2 * lane);
        const float  w  = s_scores[p];
        acc.x += w * vv.x;
        acc.y += w * vv.y;
    }
    s_part[wave][2 * lane]     = acc.x;
    s_part[wave][2 * lane + 1] = acc.y;
    __syncthreads();

    if (tid < HDIM) {
        float s = 0.0f;
#pragma unroll
        for (int w = 0; w < 8; ++w) s += s_part[w][tid];
        ctx_ws[(size_t)bh * HDIM + tid] = s * inv;   // merged[b, h*64+d]
    }
}

// ---------------------------------------------------------------------------
// Kernel 3: output projection  out = ctx @ Wo.T + bo
// ---------------------------------------------------------------------------
__global__ void __launch_bounds__(32)
out_gemm_wmma(const float* __restrict__ ctx,
              const float* __restrict__ Wo, const float* __restrict__ bo,
              float* __restrict__ out)
{
    const int lane = threadIdx.x;
    const int n0   = blockIdx.x * 16;
    const int m0   = blockIdx.y * 16;

    const v8f c = wmma_tile_xwt(ctx, Wo, m0, n0, EDIM, lane);

    const int   ncol  = n0 + (lane & 15);
    const int   mbase = m0 + ((lane < 16) ? 0 : 8);
    const float bias  = bo[ncol];
#pragma unroll
    for (int r = 0; r < 8; ++r)
        out[(size_t)(mbase + r) * EDIM + ncol] = c[r] + bias;
}

// ---------------------------------------------------------------------------
// Host launcher
// ---------------------------------------------------------------------------
extern "C" void kernel_launch(void* const* d_in, const int* in_sizes, int n_in,
                              void* d_out, int out_size, void* d_ws, size_t ws_size,
                              hipStream_t stream)
{
    (void)in_sizes; (void)n_in; (void)out_size; (void)ws_size;

    const float* hidden = (const float*)d_in[0];   // [64,1,1280]
    const float* past_k = (const float*)d_in[1];   // [64,20,448,64]
    const float* past_v = (const float*)d_in[2];   // [64,20,448,64]
    const float* kmask  = (const float*)d_in[3];   // [448]
    const float* Wq     = (const float*)d_in[4];
    const float* bq     = (const float*)d_in[5];
    const float* Wk     = (const float*)d_in[6];
    const float* Wv     = (const float*)d_in[7];
    const float* bv     = (const float*)d_in[8];
    const float* Wo     = (const float*)d_in[9];
    const float* bo     = (const float*)d_in[10];

    float* out   = (float*)d_out;                  // [64,1,1280]
    float* new_k = out + (size_t)BATCH * EDIM;     // [64,20,1,64] flat
    float* new_v = new_k + (size_t)BATCH * EDIM;   // [64,20,1,64] flat

    float* q_ws   = (float*)d_ws;                  // [64,1280] scaled q
    float* ctx_ws = q_ws + (size_t)BATCH * EDIM;   // [64,1280] merged ctx

    // 1) QKV projection: 240 n-tiles (80 per projection) x 4 m-tiles, 1 wave each
    qkv_gemm_wmma<<<dim3(240, 4), 32, 0, stream>>>(
        hidden, Wq, bq, Wk, Wv, bv, q_ws, new_k, new_v);

    // 2) Attention over 449 keys per (b,h): 1280 blocks x 8 waves
    attn_decode<<<dim3(BATCH * HEADS), 256, 0, stream>>>(
        q_ws, past_k, past_v, kmask, new_k, new_v, ctx_ws);

    // 3) Output projection: 80 n-tiles x 4 m-tiles
    out_gemm_wmma<<<dim3(80, 4), 32, 0, stream>>>(ctx_ws, Wo, bo, out);
}
